// UWMRMoE_75222057222459
// MI455X (gfx1250) — compile-verified, matrix operand order checked by reading, not run on hardware
//
#include <hip/hip_runtime.h>
#include <hip/hip_bf16.h>
#include <math.h>

#define NTOK 4096
#define DDIM 1024
#define FDIM 2048
#define NEXP 8
#define BCOEF 0.01f
#define EPSV 1e-6f

#define KSTEP 32
#define ROWE  40            // LDS row stride in elements (80B: 16 distinct bank groups)
#define STAGE_ELEMS (64 * ROWE)

typedef __attribute__((ext_vector_type(16))) __bf16 v16bf;
typedef __attribute__((ext_vector_type(8)))  float  v8f;
typedef __attribute__((ext_vector_type(4)))  int    v4i;

typedef __attribute__((address_space(1))) v4i* gptr_b128;
typedef __attribute__((address_space(3))) v4i* lptr_b128;

// ---------- CDNA5 async global->LDS + wait ----------
__device__ __forceinline__ void async_b128(const void* gsrc, unsigned lds_off) {
#if __has_builtin(__builtin_amdgcn_global_load_async_to_lds_b128)
    __builtin_amdgcn_global_load_async_to_lds_b128(
        (gptr_b128)(unsigned long long)(size_t)gsrc,
        (lptr_b128)(unsigned long long)lds_off,
        0, 0);
#else
    asm volatile("global_load_async_to_lds_b128 %0, %1, off"
                 :: "v"(lds_off), "v"((unsigned long long)(size_t)gsrc)
                 : "memory");
#endif
}

#if __has_builtin(__builtin_amdgcn_s_wait_asynccnt)
#define WAIT_ASYNC() __builtin_amdgcn_s_wait_asynccnt(0)
#else
#define WAIT_ASYNC() asm volatile("s_wait_asynccnt 0x0" ::: "memory")
#endif

// ---------- helpers ----------
__device__ __forceinline__ __bf16 f2bf(float f) {
    union { float f; unsigned u; } v; v.f = f;
    unsigned r = v.u + 0x7fffu + ((v.u >> 16) & 1u);   // round-to-nearest-even
    unsigned short s = (unsigned short)(r >> 16);
    union { unsigned short s; __bf16 b; } o; o.s = s;
    return o.b;
}

// Fill a v16bf operand per the CDNA5 16-bit A/B layout.
// Caller pre-offsets p by (half*8); we grab elements [0..7] and [16..23].
__device__ __forceinline__ v16bf load_bf16_operand(const __bf16* __restrict__ p) {
    union { uint4 q; __bf16 h[8]; } lo, hi;
    lo.q = *(const uint4*)(p);
    hi.q = *(const uint4*)(p + 16);
    v16bf a;
#pragma unroll
    for (int i = 0; i < 8; i++) { a[i] = lo.h[i]; a[i + 8] = hi.h[i]; }
    return a;
}

// ---------- fp32 -> bf16 convert ----------
__global__ void cvt_bf16_kernel(const float* __restrict__ src,
                                __bf16* __restrict__ dst, int n) {
    int i = blockIdx.x * blockDim.x + threadIdx.x;
    if (i < n) dst[i] = f2bf(src[i]);
}

// ---------- router ----------
__global__ void __launch_bounds__(256) router_kernel(
    const float* __restrict__ x, const float* __restrict__ U,
    const float* __restrict__ rw, const float* __restrict__ sb,
    const float* __restrict__ gb, float* __restrict__ combine,
    float* __restrict__ psum) {
    __shared__ float w[NEXP * DDIM];
    __shared__ float ssum[NEXP];
    int tid = threadIdx.x;
    for (int i = tid; i < NEXP * DDIM; i += blockDim.x) w[i] = rw[i];
    if (tid < NEXP) ssum[tid] = 0.f;
    __syncthreads();

    int n = blockIdx.x * blockDim.x + tid;
    const float* xr = x + (size_t)n * DDIM;
    float lg[NEXP];
#pragma unroll
    for (int e = 0; e < NEXP; e++) lg[e] = 0.f;
    for (int d = 0; d < DDIM; d++) {
        float xv = xr[d];
#pragma unroll
        for (int e = 0; e < NEXP; e++) lg[e] = fmaf(xv, w[e * DDIM + d], lg[e]);
    }
    float u = fminf(fmaxf(U[n], 0.f), 1.f);
    float mx = -1e30f;
#pragma unroll
    for (int e = 0; e < NEXP; e++) {
        lg[e] += u * sb[e] + (1.f - u) * gb[e];
        mx = fmaxf(mx, lg[e]);
    }
    float s = 0.f, p[NEXP];
#pragma unroll
    for (int e = 0; e < NEXP; e++) { p[e] = expf(lg[e] - mx); s += p[e]; }
    float inv = 1.f / s;
#pragma unroll
    for (int e = 0; e < NEXP; e++) p[e] *= inv;
#pragma unroll
    for (int e = 0; e < NEXP; e++) atomicAdd(&ssum[e], p[e]);

    int i0 = 0; float p0 = p[0];
#pragma unroll
    for (int e = 1; e < NEXP; e++) if (p[e] > p0) { p0 = p[e]; i0 = e; }
    int i1 = -1; float p1 = -1e30f;
#pragma unroll
    for (int e = 0; e < NEXP; e++) if (e != i0 && p[e] > p1) { p1 = p[e]; i1 = e; }
#pragma unroll
    for (int e = 0; e < NEXP; e++)
        combine[(size_t)n * NEXP + e] = (e == i0) ? p0 : ((e == i1) ? p1 : 0.f);

    __syncthreads();
    if (tid < NEXP) atomicAdd(&psum[tid], ssum[tid]);
}

__global__ void bal_kernel(const float* __restrict__ psum, float* __restrict__ out_scalar) {
    if (threadIdx.x == 0) {
        float s = 0.f;
        for (int e = 0; e < NEXP; e++) {
            float m = psum[e] * (1.f / (float)NTOK);
            s += m * m;
        }
        *out_scalar = (float)NEXP * s * BCOEF;
    }
}

// ---------- h = silu(x@Wg^T) * (x@Wu^T), async-LDS double-buffered ----------
// block = 128 thr (4 waves); block tile 64(M)x64(F); wave tile 16x64.
__global__ void __launch_bounds__(128) gemm_h_kernel(
    const __bf16* __restrict__ X,   // [NTOK, DDIM]
    const __bf16* __restrict__ Wg,  // [FDIM, DDIM]
    const __bf16* __restrict__ Wu,  // [FDIM, DDIM]
    __bf16* __restrict__ H) {       // [NTOK, FDIM]
    __shared__ __bf16 sA[2 * STAGE_ELEMS];
    __shared__ __bf16 sG[2 * STAGE_ELEMS];
    __shared__ __bf16 sU[2 * STAGE_ELEMS];

    const int tid  = threadIdx.x;
    const int wave = tid >> 5;
    const int lane = tid & 31;
    const int half = lane >> 4;
    const int lrow = lane & 15;
    const int m0 = blockIdx.x * 64;
    const int f0 = blockIdx.y * 64;

    const unsigned aBase = (unsigned)(size_t)&sA[0];
    const unsigned gBase = (unsigned)(size_t)&sG[0];
    const unsigned uBase = (unsigned)(size_t)&sU[0];
    const unsigned stageB = STAGE_ELEMS * 2;  // bytes per stage

    // per-thread DMA mapping: 2 x 16B chunks per panel per stage
    const int lin0 = tid, lin1 = 128 + tid;
    const int r0 = lin0 >> 2, c0 = (lin0 & 3);   // row, 16B-chunk within 64B of data
    const int r1 = lin1 >> 2, c1 = (lin1 & 3);

    auto prefetch = [&](int s, int k0) {
        int buf = s & 1;
        unsigned l0 = buf * stageB + (unsigned)(r0 * (ROWE * 2) + c0 * 16);
        unsigned l1 = buf * stageB + (unsigned)(r1 * (ROWE * 2) + c1 * 16);
        async_b128(X  + (size_t)(m0 + r0) * DDIM + k0 + c0 * 8, aBase + l0);
        async_b128(X  + (size_t)(m0 + r1) * DDIM + k0 + c1 * 8, aBase + l1);
        async_b128(Wg + (size_t)(f0 + r0) * DDIM + k0 + c0 * 8, gBase + l0);
        async_b128(Wg + (size_t)(f0 + r1) * DDIM + k0 + c1 * 8, gBase + l1);
        async_b128(Wu + (size_t)(f0 + r0) * DDIM + k0 + c0 * 8, uBase + l0);
        async_b128(Wu + (size_t)(f0 + r1) * DDIM + k0 + c1 * 8, uBase + l1);
    };

    v8f accg[4] = {}; v8f accu[4] = {};
    const int NSTAGE = DDIM / KSTEP;

    prefetch(0, 0);
    WAIT_ASYNC();
    __syncthreads();

    for (int s = 0; s < NSTAGE; s++) {
        if (s + 1 < NSTAGE) prefetch(s + 1, (s + 1) * KSTEP);
        const int buf = (s & 1) * STAGE_ELEMS;
        v16bf a = load_bf16_operand(&sA[buf + (wave * 16 + lrow) * ROWE + half * 8]);
#pragma unroll
        for (int j = 0; j < 4; j++) {
            v16bf bg = load_bf16_operand(&sG[buf + (j * 16 + lrow) * ROWE + half * 8]);
            v16bf bu = load_bf16_operand(&sU[buf + (j * 16 + lrow) * ROWE + half * 8]);
            accg[j] = __builtin_amdgcn_wmma_f32_16x16x32_bf16(
                false, a, false, bg, (short)0, accg[j], false, false);
            accu[j] = __builtin_amdgcn_wmma_f32_16x16x32_bf16(
                false, a, false, bu, (short)0, accu[j], false, false);
        }
        WAIT_ASYNC();
        __syncthreads();
    }

#pragma unroll
    for (int j = 0; j < 4; j++) {
#pragma unroll
        for (int i = 0; i < 8; i++) {
            float g = accg[j][i];
            float hv = (g / (1.f + __expf(-g))) * accu[j][i];
            int m = m0 + wave * 16 + half * 8 + i;
            int f = f0 + j * 16 + lrow;
            H[(size_t)m * FDIM + f] = f2bf(hv);
        }
    }
}

// ---------- out += combine * (h @ Wd^T), async-LDS double-buffered ----------
__global__ void __launch_bounds__(128) gemm_y_kernel(
    const __bf16* __restrict__ H,   // [NTOK, FDIM]
    const __bf16* __restrict__ Wd,  // [DDIM, FDIM]
    const float* __restrict__ cw,   // combine column (stride NEXP) or nullptr
    float* __restrict__ acc_out) {  // [NTOK, DDIM] fp32
    __shared__ __bf16 sA[2 * STAGE_ELEMS];
    __shared__ __bf16 sB[2 * STAGE_ELEMS];

    const int tid  = threadIdx.x;
    const int wave = tid >> 5;
    const int lane = tid & 31;
    const int half = lane >> 4;
    const int lrow = lane & 15;
    const int m0 = blockIdx.x * 64;
    const int d0 = blockIdx.y * 64;

    const unsigned aBase = (unsigned)(size_t)&sA[0];
    const unsigned bBase = (unsigned)(size_t)&sB[0];
    const unsigned stageB = STAGE_ELEMS * 2;

    const int lin0 = tid, lin1 = 128 + tid;
    const int r0 = lin0 >> 2, c0 = (lin0 & 3);
    const int r1 = lin1 >> 2, c1 = (lin1 & 3);

    auto prefetch = [&](int s, int k0) {
        int buf = s & 1;
        unsigned l0 = buf * stageB + (unsigned)(r0 * (ROWE * 2) + c0 * 16);
        unsigned l1 = buf * stageB + (unsigned)(r1 * (ROWE * 2) + c1 * 16);
        async_b128(H  + (size_t)(m0 + r0) * FDIM + k0 + c0 * 8, aBase + l0);
        async_b128(H  + (size_t)(m0 + r1) * FDIM + k0 + c1 * 8, aBase + l1);
        async_b128(Wd + (size_t)(d0 + r0) * FDIM + k0 + c0 * 8, bBase + l0);
        async_b128(Wd + (size_t)(d0 + r1) * FDIM + k0 + c1 * 8, bBase + l1);
    };

    v8f acc[4] = {};
    const int NSTAGE = FDIM / KSTEP;

    prefetch(0, 0);
    WAIT_ASYNC();
    __syncthreads();

    for (int s = 0; s < NSTAGE; s++) {
        if (s + 1 < NSTAGE) prefetch(s + 1, (s + 1) * KSTEP);
        const int buf = (s & 1) * STAGE_ELEMS;
        v16bf a = load_bf16_operand(&sA[buf + (wave * 16 + lrow) * ROWE + half * 8]);
#pragma unroll
        for (int j = 0; j < 4; j++) {
            v16bf b = load_bf16_operand(&sB[buf + (j * 16 + lrow) * ROWE + half * 8]);
            acc[j] = __builtin_amdgcn_wmma_f32_16x16x32_bf16(
                false, a, false, b, (short)0, acc[j], false, false);
        }
        WAIT_ASYNC();
        __syncthreads();
    }

#pragma unroll
    for (int j = 0; j < 4; j++) {
#pragma unroll
        for (int i = 0; i < 8; i++) {
            int m = m0 + wave * 16 + half * 8 + i;
            int d = d0 + j * 16 + lrow;
            float w = cw ? cw[(size_t)m * NEXP] : 1.f;
            acc_out[(size_t)m * DDIM + d] += w * acc[j][i];
        }
    }
}

// ---------- RMSNorm finalize ----------
__global__ void __launch_bounds__(256) rms_kernel(
    const float* __restrict__ acc, const float* __restrict__ nw,
    float* __restrict__ out) {
    int n = blockIdx.x, tid = threadIdx.x;
    const float* r = acc + (size_t)n * DDIM;
    float s = 0.f;
    for (int d = tid; d < DDIM; d += 256) s += r[d] * r[d];
    __shared__ float red[256];
    red[tid] = s; __syncthreads();
    for (int o = 128; o > 0; o >>= 1) {
        if (tid < o) red[tid] += red[tid + o];
        __syncthreads();
    }
    float scale = rsqrtf(red[0] * (1.f / (float)DDIM) + EPSV);
    for (int d = tid; d < DDIM; d += 256)
        out[(size_t)n * DDIM + d] = r[d] * scale * nw[d];
}

// ---------- launch ----------
extern "C" void kernel_launch(void* const* d_in, const int* in_sizes, int n_in,
                              void* d_out, int out_size, void* d_ws, size_t ws_size,
                              hipStream_t stream) {
    (void)in_sizes; (void)n_in; (void)out_size; (void)ws_size;
    const float* x  = (const float*)d_in[0];
    const float* U  = (const float*)d_in[1];
    const float* rw = (const float*)d_in[2];
    const float* sb = (const float*)d_in[3];
    const float* gb = (const float*)d_in[4];
    const float* Wg = (const float*)d_in[5];
    const float* Wu = (const float*)d_in[6];
    const float* Wd = (const float*)d_in[7];
    const float* Sg = (const float*)d_in[8];
    const float* Su = (const float*)d_in[9];
    const float* Sd = (const float*)d_in[10];
    const float* nw = (const float*)d_in[11];
    float* out = (float*)d_out;

    char* ws = (char*)d_ws;
    size_t off = 0;
    __bf16* xb  = (__bf16*)(ws + off); off += (size_t)NTOK * DDIM * 2;
    __bf16* wgb = (__bf16*)(ws + off); off += (size_t)FDIM * DDIM * 2;
    __bf16* wub = (__bf16*)(ws + off); off += (size_t)FDIM * DDIM * 2;
    __bf16* wdb = (__bf16*)(ws + off); off += (size_t)DDIM * FDIM * 2;
    __bf16* hb  = (__bf16*)(ws + off); off += (size_t)NTOK * FDIM * 2;
    float*  acc = (float*)(ws + off);  off += (size_t)NTOK * DDIM * 4;
    float*  comb= (float*)(ws + off);  off += (size_t)NTOK * NEXP * 4;
    float*  psum= (float*)(ws + off);  off += 256;

    (void)hipMemsetAsync(acc, 0, (size_t)NTOK * DDIM * sizeof(float), stream);
    (void)hipMemsetAsync(psum, 0, 256, stream);

    cvt_bf16_kernel<<<(NTOK * DDIM + 255) / 256, 256, 0, stream>>>(x, xb, NTOK * DDIM);
    router_kernel<<<NTOK / 256, 256, 0, stream>>>(x, U, rw, sb, gb, comb, psum);
    bal_kernel<<<1, 32, 0, stream>>>(psum, out + (size_t)NTOK * DDIM);

    for (int e = 0; e < NEXP + 1; e++) {
        const float* wg_src = (e < NEXP) ? (Wg + (size_t)e * FDIM * DDIM) : Sg;
        const float* wu_src = (e < NEXP) ? (Wu + (size_t)e * FDIM * DDIM) : Su;
        const float* wd_src = (e < NEXP) ? (Wd + (size_t)e * DDIM * FDIM) : Sd;
        cvt_bf16_kernel<<<(FDIM * DDIM + 255) / 256, 256, 0, stream>>>(wg_src, wgb, FDIM * DDIM);
        cvt_bf16_kernel<<<(FDIM * DDIM + 255) / 256, 256, 0, stream>>>(wu_src, wub, FDIM * DDIM);
        cvt_bf16_kernel<<<(DDIM * FDIM + 255) / 256, 256, 0, stream>>>(wd_src, wdb, DDIM * FDIM);
        gemm_h_kernel<<<dim3(NTOK / 64, FDIM / 64), 128, 0, stream>>>(xb, wgb, wub, hb);
        const float* cwp = (e < NEXP) ? (comb + e) : nullptr;
        gemm_y_kernel<<<dim3(NTOK / 64, DDIM / 64), 128, 0, stream>>>(hb, wdb, cwp, acc);
    }
    rms_kernel<<<NTOK, 256, 0, stream>>>(acc, nw, out);
}
